// TransformerEncoder_60816736911586
// MI455X (gfx1250) — matrix operand order repaired and checked by reading, not compiled, not run on hardware
//
#include <hip/hip_runtime.h>
#include <hip/hip_bf16.h>

typedef __attribute__((ext_vector_type(16))) _Float16 v16h;
typedef __attribute__((ext_vector_type(8)))  float    v8f;

#define TPB  128
#define NB   2       // batch elements per block (shared weight fragments)
#define NP   16      // tokens (P)
#define DD   49      // model dim
#define DP   64      // padded dim
#define NH   8
#define NL   6
#define FF   196
#define NCLS 10

// ---- pre-swizzled f16 weight arena in d_ws (offsets in halves) -----------
#define T2048    2048
#define OFF_PP   0
#define OFF_WQ   (OFF_PP + 2 * T2048)
#define OFF_WK   (OFF_WQ + 96 * T2048)
#define OFF_WV   (OFF_WK + 96 * T2048)
#define OFF_WH   (OFF_WV + 96 * T2048)
#define OFF_WO   (OFF_WH + 96 * T2048)
#define OFF_F1   (OFF_WO + 96 * T2048)
#define OFF_F2   (OFF_F1 + 48 * T2048)
#define OFF_M1   (OFF_F2 + 48 * T2048)
#define OFF_M2   (OFF_M1 + 2 * T2048)
#define NPREP_BLOCKS 582                  // arena = 582*2048*2 B ~= 2.28 MB

// ---- per-batch LDS layout (halves / floats) ------------------------------
// f16 arena per batch: hA | yA | mid (aliases qA/aoA/hoA/pA) | k16 (aliases tA) | vT
#define F16_PER  9216
#define O_HA     0
#define O_YA     1024
#define O_MID    2048     // 4096 halves, FFN mid (K=256)
#define O_QA     2048     // alias: attention scratch inside mid region
#define O_AOA    3072
#define O_HOA    4096
#define O_PA     5120     // 512 halves
#define O_K16    6144     // 1024 halves, K row-major
#define O_TA     6144     // alias: x-stage / mlp-mid (attention-free lifetime)
#define O_VT     7168     // 2048 halves, V transposed [dim][token]
// f32 arena per batch: h32 | y32 | fo32 | s32
#define F32_PER  3328
#define O_H32    0
#define O_Y32    1024
#define O_FO     2048
#define O_S32    3072

__device__ __forceinline__ v8f wmma_acc(v16h a, v16h b, v8f c) {
  return __builtin_amdgcn_wmma_f32_16x16x32_f16(false, a, false, b, (short)0, c, false, false);
}
__device__ __forceinline__ v8f splat8(float v) {
  v8f r = {v, v, v, v, v, v, v, v};
  return r;
}

// A-operand frag index: element (m, k) -> kc*512 + lane*16 + e
__device__ __forceinline__ int afidx(int m, int k) {
  int kc = k >> 5, kl = k & 31;
  int ln = m + ((kl >> 4) << 4);
  int e  = kl & 15;
  return (kc << 9) + (ln << 4) + e;
}
// B-operand frag index within one 2048-half tile: element (kl, n)
__device__ __forceinline__ int bfidx(int kl, int n) {
  int nt = n >> 4;
  int ln = (n & 15) + ((kl >> 4) << 4);
  return (nt << 9) + (ln << 4) + (kl & 15);
}

// Dual-batch GEMM: one B fragment (global, frag-ready) feeds two WMMAs.
__device__ __forceinline__ void mm64g2(const _Float16* __restrict__ Wf, int kChunks,
                                       const _Float16* __restrict__ A0,
                                       const _Float16* __restrict__ A1,
                                       int tid, v8f& acc0, v8f& acc1) {
  const int lane = tid & 31;
  const int wv   = tid >> 5;
  for (int kc = 0; kc < kChunks; ++kc) {
    if (kc + 1 < kChunks)
      __builtin_prefetch(Wf + ((kc + 1) << 11) + (wv << 9) + (lane << 4), 0, 1);
    v16h b  = *(const v16h*)(Wf + (kc << 11) + (wv << 9) + (lane << 4)); // 2x global b128
    v16h a0 = *(const v16h*)(A0 + (kc << 9) + (lane << 4));              // 2x ds b128
    v16h a1 = *(const v16h*)(A1 + (kc << 9) + (lane << 4));
    acc0 = wmma_acc(a0, b, acc0);
    acc1 = wmma_acc(a1, b, acc1);
  }
}

// Write both C tiles into frag-ready A buffers (bias already inside acc).
__device__ __forceinline__ void store_afrag2(v8f a0, v8f a1, _Float16* A0d, _Float16* A1d,
                                             int valid, int tid) {
  int lane = tid & 31, wv = tid >> 5;
  int c = wv * 16 + (lane & 15);
  bool ok = c < valid;
#pragma unroll
  for (int r = 0; r < 8; ++r) {
    int m = ((lane >> 4) << 3) + r;
    A0d[afidx(m, c)] = (_Float16)(ok ? a0[r] : 0.f);
    A1d[afidx(m, c)] = (_Float16)(ok ? a1[r] : 0.f);
  }
}

__device__ __forceinline__ float gelu_exact(float x) {
  return 0.5f * x * (1.0f + erff(x * 0.7071067811865475f));
}

// ---------------- weight pre-swizzle (runs once per launch) ----------------
__device__ __forceinline__ void prep_tile(const float* __restrict__ W, int rows, int cols,
                                          int ld, int ncol0, int kc,
                                          _Float16* __restrict__ dst, int tid) {
  for (int i = tid; i < 2048; i += TPB) {
    int r = i >> 6, c = i & 63;
    int gk = kc * 32 + r, gc = ncol0 + c;
    float v = (gk < rows && gc < cols) ? W[(size_t)gk * ld + gc] : 0.f;
    dst[bfidx(r, c)] = (_Float16)v;
  }
}

extern "C" __global__ void __launch_bounds__(TPB)
prep_weights(const float* __restrict__ ppw,
             const float* __restrict__ Wq, const float* __restrict__ Wk,
             const float* __restrict__ Wv, const float* __restrict__ Wh,
             const float* __restrict__ Wo,
             const float* __restrict__ f1w, const float* __restrict__ f2w,
             const float* __restrict__ m1w, const float* __restrict__ m2w,
             _Float16* __restrict__ wf) {
  int id = blockIdx.x, tid = threadIdx.x;
  if (id < 2) { prep_tile(ppw, DD, DD, DD, 0, id, wf + OFF_PP + id * T2048, tid); return; }
  id -= 2;
  if (id < 96) { int m = id >> 1, kc = id & 1;
    prep_tile(Wq + (size_t)m * DD * DD, DD, DD, DD, 0, kc, wf + OFF_WQ + id * T2048, tid); return; }
  id -= 96;
  if (id < 96) { int m = id >> 1, kc = id & 1;
    prep_tile(Wk + (size_t)m * DD * DD, DD, DD, DD, 0, kc, wf + OFF_WK + id * T2048, tid); return; }
  id -= 96;
  if (id < 96) { int m = id >> 1, kc = id & 1;
    prep_tile(Wv + (size_t)m * DD * DD, DD, DD, DD, 0, kc, wf + OFF_WV + id * T2048, tid); return; }
  id -= 96;
  if (id < 96) { int m = id >> 1, kc = id & 1;
    prep_tile(Wh + (size_t)m * DD * DD, DD, DD, DD, 0, kc, wf + OFF_WH + id * T2048, tid); return; }
  id -= 96;
  if (id < 96) {
    int m = id >> 1, kc = id & 1;
    int l = m >> 3, hh = m & 7;
    prep_tile(Wo + ((size_t)l * (NH * DD) + hh * DD) * DD, DD, DD, DD, 0, kc,
              wf + OFF_WO + id * T2048, tid);
    return;
  }
  id -= 96;
  if (id < 48) {
    int l = id / 8, r = id % 8, ng = r >> 1, kc = r & 1;
    prep_tile(f1w + (size_t)l * DD * FF, DD, FF, FF, ng * 64, kc,
              wf + OFF_F1 + id * T2048, tid);
    return;
  }
  id -= 48;
  if (id < 48) {
    int l = id / 8, kc = id % 8;
    prep_tile(f2w + (size_t)l * FF * DD, FF, DD, DD, 0, kc, wf + OFF_F2 + id * T2048, tid);
    return;
  }
  id -= 48;
  if (id < 2) { prep_tile(m1w, DD, DD, DD, 0, id, wf + OFF_M1 + id * T2048, tid); return; }
  id -= 2;
  prep_tile(m2w, DD, DD, DD, 0, id, wf + OFF_M2 + id * T2048, tid);
}

// ------------------------------ main kernel -------------------------------
extern "C" __global__ void __launch_bounds__(TPB)
vit_fwd(const float* __restrict__ x, const int* __restrict__ trg,
        const float* __restrict__ ppb, const float* __restrict__ pos,
        const float* __restrict__ bq, const float* __restrict__ bk,
        const float* __restrict__ bv, const float* __restrict__ bh,
        const float* __restrict__ bo,
        const float* __restrict__ ln1w, const float* __restrict__ ln1b,
        const float* __restrict__ ln2w, const float* __restrict__ ln2b,
        const float* __restrict__ f1b, const float* __restrict__ f2b,
        const float* __restrict__ m1b, const float* __restrict__ m2b,
        const float* __restrict__ c1w, const float* __restrict__ c1b,
        const float* __restrict__ c2w, const float* __restrict__ c2b,
        const _Float16* __restrict__ wf,
        float* __restrict__ out, int Bn) {
  __shared__ _Float16 f16b[NB * F16_PER];   // 36.9 KB
  __shared__ float    f32b[NB * F32_PER];   // 26.6 KB
  __shared__ float    red[256];             //  1.0 KB  (total ~63 KB)

  const int tid  = threadIdx.x;
  const int lane = tid & 31;
  const int wv   = tid >> 5;
  const int c    = wv * 16 + (lane & 15);   // output column owned by this thread
  const bool okD = c < DD;
  const int bi0  = blockIdx.x * NB;
  const float* xb0 = x + (size_t)(bi0 + 0) * NP * DD;
  const float* xb1 = x + (size_t)(bi0 + 1) * NP * DD;
  _Float16* F0 = f16b;
  _Float16* F1 = f16b + F16_PER;
  float*    G0 = f32b;
  float*    G1 = f32b + F32_PER;
  const v8f vzero = splat8(0.f);

  // stage x (both batch elems) into frag-ready tA
  for (int i = tid; i < NP * DP; i += TPB) {
    int m = i >> 6, cc = i & 63;
    int fi = afidx(m, cc);
    F0[O_TA + fi] = (_Float16)((cc < DD) ? xb0[m * DD + cc] : 0.f);
    F1[O_TA + fi] = (_Float16)((cc < DD) ? xb1[m * DD + cc] : 0.f);
  }
  __syncthreads();
  {  // h = x @ patch_proj_w + b + pos_enc
    v8f a0 = splat8(okD ? ppb[c] : 0.f), a1 = a0;
    mm64g2(wf + OFF_PP, 2, F0 + O_TA, F1 + O_TA, tid, a0, a1);
#pragma unroll
    for (int r = 0; r < 8; ++r) {
      int m = ((lane >> 4) << 3) + r;
      float pz = okD ? pos[m * DD + c] : 0.f;
      float v0 = okD ? a0[r] + pz : 0.f;
      float v1 = okD ? a1[r] + pz : 0.f;
      int fi = afidx(m, c);
      G0[O_H32 + m * DP + c] = v0; F0[O_HA + fi] = (_Float16)v0;
      G1[O_H32 + m * DP + c] = v1; F1[O_HA + fi] = (_Float16)v1;
    }
  }
  __syncthreads();

  for (int l = 0; l < NL; ++l) {
    v8f at0 = vzero, at1 = vzero;  // per-wave attn_out tiles (both batches)
    for (int hh = 0; hh < NH; ++hh) {
      int mi = l * NH + hh;
      size_t bOff = (size_t)mi * DD;
      {  // Q -> qA
        v8f a0 = splat8(okD ? bq[bOff + c] : 0.f), a1 = a0;
        mm64g2(wf + OFF_WQ + (size_t)mi * 2 * T2048, 2, F0 + O_HA, F1 + O_HA, tid, a0, a1);
        store_afrag2(a0, a1, F0 + O_QA, F1 + O_QA, DD, tid);
      }
      {  // K -> k16 row-major
        v8f a0 = splat8(okD ? bk[bOff + c] : 0.f), a1 = a0;
        mm64g2(wf + OFF_WK + (size_t)mi * 2 * T2048, 2, F0 + O_HA, F1 + O_HA, tid, a0, a1);
#pragma unroll
        for (int r = 0; r < 8; ++r) {
          int m = ((lane >> 4) << 3) + r;
          F0[O_K16 + m * DP + c] = (_Float16)(okD ? a0[r] : 0.f);
          F1[O_K16 + m * DP + c] = (_Float16)(okD ? a1[r] : 0.f);
        }
      }
      {  // V -> vT transposed; zero token slots 16..31
        v8f a0 = splat8(okD ? bv[bOff + c] : 0.f), a1 = a0;
        mm64g2(wf + OFF_WV + (size_t)mi * 2 * T2048, 2, F0 + O_HA, F1 + O_HA, tid, a0, a1);
#pragma unroll
        for (int r = 0; r < 8; ++r) {
          int m = ((lane >> 4) << 3) + r;
          F0[O_VT + c * 32 + m]      = (_Float16)(okD ? a0[r] : 0.f);
          F0[O_VT + c * 32 + m + 16] = (_Float16)0.f;
          F1[O_VT + c * 32 + m]      = (_Float16)(okD ? a1[r] : 0.f);
          F1[O_VT + c * 32 + m + 16] = (_Float16)0.f;
        }
      }
      __syncthreads();  // qA, k16, vT visible
      {  // S = (Q @ K^T) * 1/7; waves 0,2 -> batch0, waves 1,3 -> batch1
        int nb = wv & 1;
        const _Float16* Fq = (nb ? F1 : F0);
        v8f a = vzero;
#pragma unroll
        for (int kc = 0; kc < 2; ++kc) {
          v16h af = *(const v16h*)(Fq + O_QA + (kc << 9) + (lane << 4));
          v16h bf = *(const v16h*)(Fq + O_K16 + (lane & 15) * DP + kc * 32 + ((lane >> 4) << 4));
          a = wmma_acc(af, bf, a);
        }
        if (wv < 2) {  // wv0 writes batch0, wv1 writes batch1
          float* S = (wv ? G1 : G0) + O_S32;
          int cc = lane & 15;
#pragma unroll
          for (int r = 0; r < 8; ++r) {
            int m = ((lane >> 4) << 3) + r;
            S[m * 16 + cc] = a[r] * 0.14285714285714285f;
          }
        }
      }
      __syncthreads();
      if (tid < 32) {  // row softmax (both batches) -> pA (frag-ready, zero-padded)
        int nb = tid >> 4, m = tid & 15;
        float* S = (nb ? G1 : G0) + O_S32;
        _Float16* P = (nb ? F1 : F0) + O_PA;
        float mx = -1e30f;
        for (int cc = 0; cc < 16; ++cc) mx = fmaxf(mx, S[m * 16 + cc]);
        float s = 0.f;
        for (int cc = 0; cc < 16; ++cc) {
          float e = __expf(S[m * 16 + cc] - mx);
          S[m * 16 + cc] = e;
          s += e;
        }
        float inv = 1.f / s;
        for (int k = 0; k < 8; ++k) {
          P[m * 16 + k]            = (_Float16)(S[m * 16 + k] * inv);
          P[m * 16 + 8 + k]        = (_Float16)(S[m * 16 + 8 + k] * inv);
          P[(m + 16) * 16 + k]     = (_Float16)0.f;
          P[(m + 16) * 16 + 8 + k] = (_Float16)0.f;
        }
      }
      __syncthreads();
#pragma unroll
      for (int nb = 0; nb < NB; ++nb) {  // ao = attn @ V -> aoA (both batches)
        _Float16* F = nb ? F1 : F0;
        v16h af = *(const v16h*)(F + O_PA + (lane << 4));
        v16h bf = *(const v16h*)(F + O_VT + (wv * 16 + (lane & 15)) * 32 + ((lane >> 4) << 4));
        v8f a = wmma_acc(af, bf, vzero);
#pragma unroll
        for (int r = 0; r < 8; ++r) {
          int m = ((lane >> 4) << 3) + r;
          F[O_AOA + afidx(m, c)] = (_Float16)(okD ? a[r] : 0.f);
        }
      }
      __syncthreads();  // aoA visible
      {  // ho = ao @ Wh + bh -> hoA
        v8f a0 = splat8(okD ? bh[bOff + c] : 0.f), a1 = a0;
        mm64g2(wf + OFF_WH + (size_t)mi * 2 * T2048, 2, F0 + O_AOA, F1 + O_AOA, tid, a0, a1);
        store_afrag2(a0, a1, F0 + O_HOA, F1 + O_HOA, DD, tid);
      }
      __syncthreads();  // hoA visible
      // attn_out += ho @ Wo[hh*49:(hh+1)*49, :]
      mm64g2(wf + OFF_WO + (size_t)mi * 2 * T2048, 2, F0 + O_HOA, F1 + O_HOA, tid, at0, at1);
      // next-head hoA/qA/k16/vT writes are separated from these reads by the QKV sync
    }
    {  // fo32 = attn_out + bo
      float bb = okD ? bo[l * DD + c] : 0.f;
#pragma unroll
      for (int r = 0; r < 8; ++r) {
        int m = ((lane >> 4) << 3) + r;
        G0[O_FO + m * DP + c] = okD ? at0[r] + bb : 0.f;
        G1[O_FO + m * DP + c] = okD ? at1[r] + bb : 0.f;
      }
    }
    __syncthreads();
    if (tid < 32) {  // y = LN(h + attn_out), both batches
      int nb = tid >> 4, m = tid & 15;
      float* G = nb ? G1 : G0;
      _Float16* F = nb ? F1 : F0;
      float mu = 0.f;
      for (int cc = 0; cc < DD; ++cc) mu += G[O_H32 + m * DP + cc] + G[O_FO + m * DP + cc];
      mu *= (1.f / DD);
      float var = 0.f;
      for (int cc = 0; cc < DD; ++cc) {
        float d = G[O_H32 + m * DP + cc] + G[O_FO + m * DP + cc] - mu;
        var += d * d;
      }
      var *= (1.f / DD);
      float rs = rsqrtf(var + 1e-5f);
      for (int cc = 0; cc < DP; ++cc) {
        float o = 0.f;
        if (cc < DD) {
          float v = G[O_H32 + m * DP + cc] + G[O_FO + m * DP + cc];
          o = (v - mu) * rs * ln1w[l * DD + cc] + ln1b[l * DD + cc];
        }
        G[O_Y32 + m * DP + cc] = o;
        F[O_YA + afidx(m, cc)] = (_Float16)o;
      }
    }
    __syncthreads();  // yA visible
    for (int ng = 0; ng < 4; ++ng) {  // mid = gelu(y @ ffn1 + b1); disjoint col groups
      int gc = ng * 64 + c;
      bool okF = gc < FF;
      v8f a0 = splat8(okF ? f1b[l * FF + gc] : 0.f), a1 = a0;
      mm64g2(wf + OFF_F1 + (size_t)(l * 4 + ng) * 2 * T2048, 2, F0 + O_YA, F1 + O_YA,
             tid, a0, a1);
#pragma unroll
      for (int r = 0; r < 8; ++r) {
        int m = ((lane >> 4) << 3) + r;
        int fi = afidx(m, gc);
        F0[O_MID + fi] = (_Float16)(okF ? gelu_exact(a0[r]) : 0.f);
        F1[O_MID + fi] = (_Float16)(okF ? gelu_exact(a1[r]) : 0.f);
      }
    }
    __syncthreads();  // midA visible
    {  // ffn_out = mid @ ffn2 + b2 -> fo32
      v8f a0 = splat8(okD ? f2b[l * DD + c] : 0.f), a1 = a0;
      mm64g2(wf + OFF_F2 + (size_t)l * 8 * T2048, 8, F0 + O_MID, F1 + O_MID, tid, a0, a1);
#pragma unroll
      for (int r = 0; r < 8; ++r) {
        int m = ((lane >> 4) << 3) + r;
        G0[O_FO + m * DP + c] = okD ? a0[r] : 0.f;
        G1[O_FO + m * DP + c] = okD ? a1[r] : 0.f;
      }
    }
    __syncthreads();
    if (tid < 32) {  // h = LN(y + ffn), both batches
      int nb = tid >> 4, m = tid & 15;
      float* G = nb ? G1 : G0;
      _Float16* F = nb ? F1 : F0;
      float mu = 0.f;
      for (int cc = 0; cc < DD; ++cc) mu += G[O_Y32 + m * DP + cc] + G[O_FO + m * DP + cc];
      mu *= (1.f / DD);
      float var = 0.f;
      for (int cc = 0; cc < DD; ++cc) {
        float d = G[O_Y32 + m * DP + cc] + G[O_FO + m * DP + cc] - mu;
        var += d * d;
      }
      var *= (1.f / DD);
      float rs = rsqrtf(var + 1e-5f);
      for (int cc = 0; cc < DP; ++cc) {
        float o = 0.f;
        if (cc < DD) {
          float v = G[O_Y32 + m * DP + cc] + G[O_FO + m * DP + cc];
          o = (v - mu) * rs * ln2w[l * DD + cc] + ln2b[l * DD + cc];
        }
        G[O_H32 + m * DP + cc] = o;
        F[O_HA + afidx(m, cc)] = (_Float16)o;
      }
    }
    __syncthreads();  // hA visible
    {  // t = h @ mlp_w1 + b1 -> tA (aliases k16; attention reads long finished)
      v8f a0 = splat8(okD ? m1b[c] : 0.f), a1 = a0;
      mm64g2(wf + OFF_M1, 2, F0 + O_HA, F1 + O_HA, tid, a0, a1);
      store_afrag2(a0, a1, F0 + O_TA, F1 + O_TA, DD, tid);
    }
    __syncthreads();  // tA visible
    {  // h = t @ mlp_w2 + b2
      v8f a0 = splat8(okD ? m2b[c] : 0.f), a1 = a0;
      mm64g2(wf + OFF_M2, 2, F0 + O_TA, F1 + O_TA, tid, a0, a1);
#pragma unroll
      for (int r = 0; r < 8; ++r) {
        int m = ((lane >> 4) << 3) + r;
        float v0 = okD ? a0[r] : 0.f;
        float v1 = okD ? a1[r] : 0.f;
        int fi = afidx(m, c);
        G0[O_H32 + m * DP + c] = v0; F0[O_HA + fi] = (_Float16)v0;
        G1[O_H32 + m * DP + c] = v1; F1[O_HA + fi] = (_Float16)v1;
      }
    }
    __syncthreads();  // hA visible for next layer
  }

  // ------------ pool + classifier + loss/acc (two thread-halves) ------------
  {
    int t  = tid & 63;
    int nb = tid >> 6;
    const float* H = (nb ? G1 : G0) + O_H32;
    float* rd = red + nb * 128;
    if (t < DD) {
      float s = 0.f;
      for (int m = 0; m < NP; ++m) s += H[m * DP + t];
      rd[t] = s * (1.f / NP);
    }
    __syncthreads();
    if (t < 25) {
      float s = c1b[t];
      for (int cc = 0; cc < DD; ++cc) s += rd[cc] * c1w[cc * 25 + t];
      rd[64 + t] = s;
    }
    __syncthreads();
    if (t < NCLS) {
      float s = c2b[t];
      for (int cc = 0; cc < 25; ++cc) s += rd[64 + cc] * c2w[cc * NCLS + t];
      rd[96 + t] = s;
    }
    __syncthreads();
    if (t == 0) {
      float mx = -1e30f;
      int am = 0;
      for (int j = 0; j < NCLS; ++j) {
        float v = rd[96 + j];
        if (v > mx) { mx = v; am = j; }
      }
      float s = 0.f;
      for (int j = 0; j < NCLS; ++j) s += __expf(rd[96 + j] - mx);
      float lse = mx + logf(s);
      int tg = trg[bi0 + nb];
      float loss = -(rd[96 + tg] - lse);
      atomicAdd(&out[0], loss / (float)Bn);
      atomicAdd(&out[1], (am == tg) ? (1.f / (float)Bn) : 0.f);
    }
  }
}

extern "C" __global__ void zero_out_kernel(float* out) {
  if (threadIdx.x < 2) out[threadIdx.x] = 0.f;
}

extern "C" void kernel_launch(void* const* d_in, const int* in_sizes, int n_in,
                              void* d_out, int out_size, void* d_ws, size_t ws_size,
                              hipStream_t stream) {
  (void)n_in; (void)out_size; (void)ws_size;
  const float* x    = (const float*)d_in[0];
  const int*   trg  = (const int*)d_in[1];
  const float* ppw  = (const float*)d_in[2];
  const float* ppb  = (const float*)d_in[3];
  const float* pos  = (const float*)d_in[4];
  const float* Wq   = (const float*)d_in[5];
  const float* bq   = (const float*)d_in[6];
  const float* Wk   = (const float*)d_in[7];
  const float* bk   = (const float*)d_in[8];
  const float* Wv   = (const float*)d_in[9];
  const float* bv   = (const float*)d_in[10];
  const float* Wh   = (const float*)d_in[11];
  const float* bh   = (const float*)d_in[12];
  const float* Wo   = (const float*)d_in[13];
  const float* bo   = (const float*)d_in[14];
  const float* ln1w = (const float*)d_in[15];
  const float* ln1b = (const float*)d_in[16];
  const float* ln2w = (const float*)d_in[17];
  const float* ln2b = (const float*)d_in[18];
  const float* f1w  = (const float*)d_in[19];
  const float* f1b  = (const float*)d_in[20];
  const float* f2w  = (const float*)d_in[21];
  const float* f2b  = (const float*)d_in[22];
  const float* m1w  = (const float*)d_in[23];
  const float* m1b  = (const float*)d_in[24];
  const float* m2w  = (const float*)d_in[25];
  const float* m2b  = (const float*)d_in[26];
  const float* c1w  = (const float*)d_in[27];
  const float* c1b  = (const float*)d_in[28];
  const float* c2w  = (const float*)d_in[29];
  const float* c2b  = (const float*)d_in[30];

  int Bn = in_sizes[0] / (NP * DD);
  float* out = (float*)d_out;
  _Float16* wf = (_Float16*)d_ws;  // needs 582*2048*2 B ~= 2.28 MB of scratch

  prep_weights<<<NPREP_BLOCKS, TPB, 0, stream>>>(ppw, Wq, Wk, Wv, Wh, Wo,
                                                 f1w, f2w, m1w, m2w, wf);
  zero_out_kernel<<<1, 32, 0, stream>>>(out);
  vit_fwd<<<Bn / NB, TPB, 0, stream>>>(x, trg, ppb, pos, bq, bk, bv, bh, bo,
                                       ln1w, ln1b, ln2w, ln2b, f1b, f2b, m1b, m2b,
                                       c1w, c1b, c2w, c2b, wf, out, Bn);
}
